// DecoderLayer_31645319037344
// MI455X (gfx1250) — matrix operand order repaired and checked by reading, not compiled, not run on hardware
//
#include <hip/hip_runtime.h>

// ---------------------------------------------------------------------------
// MI455X (gfx1250, wave32) decoder layer.
// All large GEMMs use v_wmma_f32_16x16x32_bf16 with f32 accumulation.
// Data-movement design:
//   * A fragments: global -> registers directly (each lane loads exactly its
//     own WMMA fragment rows as contiguous vector loads). No LDS for A.
//   * B fragments: staged in LDS in *fragment-major* layout so each lane
//     reads 16 bf16 elements as one contiguous 32B chunk (ds_load_b128 x2),
//     double-buffered, one barrier per K-step. All 4 B fragments are loaded
//     up-front so the WMMA burst only needs partial dscnt waits.
//   * BM=128: each wave owns 32 rows (2 A fragments) x 64 cols -> 8 WMMAs
//     per K-step reusing 4 B fragments (2x arithmetic intensity vs BM=64).
//   * Weights transposed during f32->bf16 conversion so B staging is one
//     contiguous 32B global vector load per thread per K-step.
// Workspace requirement: ~420 MB.
// ---------------------------------------------------------------------------

typedef __bf16 bf16_t;
typedef __attribute__((ext_vector_type(16))) __bf16 bf16x16;
typedef __attribute__((ext_vector_type(8)))  __bf16 bf16x8;
typedef __attribute__((ext_vector_type(8)))  float  floatx8;

#define BM 128
#define BN 64
#define BK 32

// ---------------------------------------------------------------------------
// Batched WMMA GEMM:  C[z] = alpha * A[z] @ B[z] + bias  (optional ReLU)
//   A: bf16 or f32 (converted while loading), row-major, lda
//   B: bf16, row-major [K x N] (btrans=0) or [N x K] read transposed (btrans=1)
//   C: optional f32 out and/or bf16 out.
// Block: 128 threads = 4 waves. Wave w computes rows [m0+32w, m0+32w+32).
// Requires M%128==0, N%64==0, K%32==0, lda/ldb multiples of 8 (16 for btrans).
// ---------------------------------------------------------------------------
__global__ void __launch_bounds__(128)
gemm_wmma(const void* __restrict__ Aptr, int a_is_f32, long az, int lda,
          const bf16_t* __restrict__ Bptr, long bz, int ldb, int btrans,
          const float* __restrict__ bias,
          float* __restrict__ Cf, long cfz, int ldcf,
          bf16_t* __restrict__ Cb, long cbz, int ldcb,
          int M, int N, int K, float alpha, int relu)
{
    (void)M; (void)N;
    // B tile, fragment-major, double buffered: ((ntile*32 + lane)*16 + e)
    __shared__ alignas(32) bf16_t Bsh[2][BK * BN];

    const int tid  = threadIdx.x;
    const int wave = tid >> 5;
    const int lane = tid & 31;
    const int z    = blockIdx.z;
    const int m0   = blockIdx.y * BM;
    const int n0   = blockIdx.x * BN;

    const bf16_t* Bg = Bptr + (long)z * bz;
    const float*  Af = a_is_f32 ? ((const float*)Aptr + (long)z * az) : nullptr;
    const bf16_t* Ab = a_is_f32 ? nullptr : ((const bf16_t*)Aptr + (long)z * az);

    // A fragment mapping: lane holds row m = rowBase + (lane&15);
    // lanes 0-15: K in {0..15}; lanes 16-31: K in {16..31}.
    const int  aRow0 = m0 + (wave << 5) + (lane & 15);        // fragment r=0
    const int  aRow1 = aRow0 + 16;                            // fragment r=1
    const int  aHi   = (lane >= 16);
    // B staging mapping: thread stages n-subtile sj, column sn, 16 contiguous K.
    const int  sj   = tid >> 5;
    const int  sn   = n0 + (sj << 4) + (lane & 15);
    const int  skb  = (lane >= 16) ? 16 : 0;

    auto loadA = [&](int row, int k0) -> bf16x16 {
        bf16x16 f;
        const int kb0 = k0 + (aHi ? 16 : 0);
        const int kb1 = k0 + (aHi ? 24 : 8);
        if (a_is_f32) {
            floatx8 v0 = *(const floatx8*)&Af[(long)row * lda + kb0];
            floatx8 v1 = *(const floatx8*)&Af[(long)row * lda + kb1];
            #pragma unroll
            for (int e = 0; e < 8; ++e) { f[e] = (bf16_t)v0[e]; f[8 + e] = (bf16_t)v1[e]; }
        } else {
            bf16x8 v0 = *(const bf16x8*)&Ab[(long)row * lda + kb0];
            bf16x8 v1 = *(const bf16x8*)&Ab[(long)row * lda + kb1];
            #pragma unroll
            for (int e = 0; e < 8; ++e) { f[e] = v0[e]; f[8 + e] = v1[e]; }
        }
        return f;
    };
    auto stageB = [&](int buf, int k0) {
        bf16x16 tmp;
        if (btrans) {
            tmp = *(const bf16x16*)&Bg[(long)sn * ldb + (k0 + skb)];
        } else {
            #pragma unroll
            for (int e = 0; e < 16; ++e)
                tmp[e] = Bg[(long)(k0 + skb + e) * ldb + sn];
        }
        *reinterpret_cast<bf16x16*>(&Bsh[buf][tid << 4]) = tmp;
    };

    floatx8 acc0[4], acc1[4];
    #pragma unroll
    for (int j = 0; j < 4; ++j)
        #pragma unroll
        for (int i = 0; i < 8; ++i) { acc0[j][i] = 0.f; acc1[j][i] = 0.f; }

    bf16x16 aCur0 = loadA(aRow0, 0);
    bf16x16 aCur1 = loadA(aRow1, 0);
    stageB(0, 0);
    int cur = 0;

    for (int k0 = 0; k0 < K; k0 += BK) {
        __syncthreads();                 // buffer `cur` staged; prev reads done

        // Pull all 4 B fragments first: 8x ds_load_b128 in flight, WMMAs can
        // start as soon as the first pair lands (partial dscnt waits).
        bf16x16 bfr[4];
        #pragma unroll
        for (int j = 0; j < 4; ++j)
            bfr[j] = *reinterpret_cast<const bf16x16*>(
                &Bsh[cur][((j << 5) | lane) << 4]);

        bf16x16 aNext0 = aCur0, aNext1 = aCur1;
        if (k0 + BK < K) {
            aNext0 = loadA(aRow0, k0 + BK);      // overlap next fetch with WMMAs
            aNext1 = loadA(aRow1, k0 + BK);
            stageB(cur ^ 1, k0 + BK);
            if ((tid & 31) == 0) {               // hint tile-after-next into L2
                long pn = btrans ? ((long)n0 * ldb + (k0 + 2 * BK))
                                 : ((long)(k0 + 2 * BK) * ldb + n0);
                __builtin_prefetch(&Bg[pn], 0, 1);
            }
        }

        #pragma unroll
        for (int j = 0; j < 4; ++j) {
            acc0[j] = __builtin_amdgcn_wmma_f32_16x16x32_bf16(
                false, aCur0, false, bfr[j], (short)0, acc0[j], false, false);
            acc1[j] = __builtin_amdgcn_wmma_f32_16x16x32_bf16(
                false, aCur1, false, bfr[j], (short)0, acc1[j], false, false);
        }
        aCur0 = aNext0;
        aCur1 = aNext1;
        cur ^= 1;
    }

    float*  Cfz = Cf ? Cf + (long)z * cfz : nullptr;
    bf16_t* Cbz = Cb ? Cb + (long)z * cbz : nullptr;
    for (int j = 0; j < 4; ++j) {
        int n = n0 + j * 16 + (lane & 15);
        for (int i = 0; i < 8; ++i) {
            int mb = m0 + (wave << 5) + i + ((lane >= 16) ? 8 : 0);
            {
                float v = acc0[j][i] * alpha;
                if (bias) v += bias[n];
                if (relu) v = fmaxf(v, 0.f);
                if (Cfz) Cfz[(long)mb * ldcf + n] = v;
                if (Cbz) Cbz[(long)mb * ldcb + n] = (bf16_t)v;
            }
            {
                int m = mb + 16;
                float v = acc1[j][i] * alpha;
                if (bias) v += bias[n];
                if (relu) v = fmaxf(v, 0.f);
                if (Cfz) Cfz[(long)m * ldcf + n] = v;
                if (Cbz) Cbz[(long)m * ldcb + n] = (bf16_t)v;
            }
        }
    }
}

// ---------------------------------------------------------------------------
// Transposing f32 -> bf16 weight conversion: out[n*K + k] = (bf16)in[k*N + n].
// ---------------------------------------------------------------------------
__global__ void __launch_bounds__(256)
cvt_f32_bf16_t(const float* __restrict__ in, bf16_t* __restrict__ out, int K, int N)
{
    __shared__ float tile[32][33];
    const int kb = blockIdx.y * 32, nb = blockIdx.x * 32;
    const int tx = threadIdx.x & 31, ty = threadIdx.x >> 5;   // 32 x 8
    for (int i = ty; i < 32; i += 8)
        tile[i][tx] = in[(long)(kb + i) * N + (nb + tx)];
    __syncthreads();
    for (int i = ty; i < 32; i += 8)
        out[(long)(nb + i) * K + (kb + tx)] = (bf16_t)tile[tx][i];
}

// LayerNorm over last dim (D), bf16 output (+ optional f32). One block per row.
__global__ void __launch_bounds__(256)
layernorm_kernel(const float* __restrict__ x, const float* __restrict__ g,
                 const float* __restrict__ b, bf16_t* __restrict__ ob,
                 float* __restrict__ of, int D)
{
    __shared__ float red[256];
    const int row = blockIdx.x;
    const float* xr = x + (long)row * D;
    float s = 0.f, ss = 0.f;
    for (int d = threadIdx.x; d < D; d += 256) { float v = xr[d]; s += v; ss += v * v; }
    red[threadIdx.x] = s; __syncthreads();
    for (int o = 128; o > 0; o >>= 1) { if (threadIdx.x < o) red[threadIdx.x] += red[threadIdx.x + o]; __syncthreads(); }
    float mean = red[0] / D; __syncthreads();
    red[threadIdx.x] = ss; __syncthreads();
    for (int o = 128; o > 0; o >>= 1) { if (threadIdx.x < o) red[threadIdx.x] += red[threadIdx.x + o]; __syncthreads(); }
    float var  = red[0] / D - mean * mean;
    float rstd = rsqrtf(var + 1e-6f);
    for (int d = threadIdx.x; d < D; d += 256) {
        float y = g[d] * (xr[d] - mean) * rstd + b[d];
        ob[(long)row * D + d] = (bf16_t)y;
        if (of) of[(long)row * D + d] = y;
    }
}

// In-place row softmax over S; causal: allowed s <= (row % Tmod).
__global__ void __launch_bounds__(256)
softmax_rows(float* __restrict__ p, int S, int Tmod, int causal)
{
    __shared__ float red[256];
    const long row = blockIdx.x;
    float* pr = p + row * (long)S;
    const int limit = causal ? (int)(row % Tmod) : S - 1;

    float m = -3.4e38f;
    for (int s = threadIdx.x; s <= limit; s += 256) m = fmaxf(m, pr[s]);
    red[threadIdx.x] = m; __syncthreads();
    for (int o = 128; o > 0; o >>= 1) { if (threadIdx.x < o) red[threadIdx.x] = fmaxf(red[threadIdx.x], red[threadIdx.x + o]); __syncthreads(); }
    m = red[0]; __syncthreads();

    float sum = 0.f;
    for (int s = threadIdx.x; s <= limit; s += 256) sum += expf(pr[s] - m);
    red[threadIdx.x] = sum; __syncthreads();
    for (int o = 128; o > 0; o >>= 1) { if (threadIdx.x < o) red[threadIdx.x] += red[threadIdx.x + o]; __syncthreads(); }
    float inv = 1.f / red[0];

    for (int s = threadIdx.x; s < S; s += 256)
        pr[s] = (s <= limit) ? expf(pr[s] - m) * inv : 0.f;
}

// gate = softmax(h @ w2 + b2), N in {3,4}; one block per row, K=1024.
__global__ void __launch_bounds__(128)
gating_head(const bf16_t* __restrict__ h, const float* __restrict__ w2,
            const float* __restrict__ b2, float* __restrict__ gate, int K, int N)
{
    __shared__ float red[128];
    __shared__ float logits[8];
    const int row = blockIdx.x;
    const bf16_t* hr = h + (long)row * K;
    for (int n = 0; n < N; ++n) {
        float p = 0.f;
        for (int k = threadIdx.x; k < K; k += 128) p += (float)hr[k] * w2[(long)k * N + n];
        red[threadIdx.x] = p; __syncthreads();
        for (int o = 64; o > 0; o >>= 1) { if (threadIdx.x < o) red[threadIdx.x] += red[threadIdx.x + o]; __syncthreads(); }
        if (threadIdx.x == 0) logits[n] = red[0] + b2[n];
        __syncthreads();
    }
    if (threadIdx.x == 0) {
        float m = -3.4e38f;
        for (int n = 0; n < N; ++n) m = fmaxf(m, logits[n]);
        float s = 0.f;
        for (int n = 0; n < N; ++n) { logits[n] = expf(logits[n] - m); s += logits[n]; }
        for (int n = 0; n < N; ++n) gate[(long)row * N + n] = logits[n] / s;
    }
}

// Sequential flow-diffusion scan: one persistent 1024-thread block per batch.
__global__ void __launch_bounds__(1024)
flow_scan(const float* __restrict__ E, const float* __restrict__ fg,
          float* __restrict__ attn3, int T, int S)
{
    __shared__ float prev[1024];
    __shared__ float t1[1024];
    __shared__ float t2[1024];
    __shared__ float red[1024];
    const int b = blockIdx.x;
    const int s = threadIdx.x;
    prev[s] = (s == 0) ? 1.f : 0.f;
    __syncthreads();
    const float* Er = E + ((long)b * S + s) * S;

    for (int t = 0; t < T; ++t) {
        const float* g = fg + ((long)b * T + t) * 4;
        const float g0 = g[0], g1 = g[1], g2 = g[2], g3 = g[3];

        float h1 = 0.f;
        for (int k = 0; k < S; ++k) h1 += Er[k] * prev[k];
        t1[s] = h1; __syncthreads();
        float h2 = 0.f;
        for (int k = 0; k < S; ++k) h2 += Er[k] * t1[k];
        t2[s] = h2; __syncthreads();
        float h3 = 0.f;
        for (int k = 0; k < S; ++k) h3 += Er[k] * t2[k];

        float sc = g0 * prev[s] + g1 * h1 + g2 * h2 + g3 * h3;
        __syncthreads();
        prev[s] = sc;

        red[s] = sc; __syncthreads();
        for (int o = 512; o > 0; o >>= 1) { if (s < o) red[s] = fmaxf(red[s], red[s + o]); __syncthreads(); }
        float m = red[0]; __syncthreads();
        float e = expf(sc - m);
        red[s] = e; __syncthreads();
        for (int o = 512; o > 0; o >>= 1) { if (s < o) red[s] += red[s + o]; __syncthreads(); }
        float inv = 1.f / red[0];
        attn3[((long)b * T + t) * S + s] = e * inv;
        __syncthreads();
    }
}

__global__ void __launch_bounds__(256)
context_combine(const float* __restrict__ x1, const float* __restrict__ c1,
                const float* __restrict__ c2, const float* __restrict__ c3,
                const float* __restrict__ ig, float* __restrict__ out, int D)
{
    const int row = blockIdx.x;
    const float g0 = ig[(long)row * 3 + 0];
    const float g1 = ig[(long)row * 3 + 1];
    const float g2 = ig[(long)row * 3 + 2];
    const long base = (long)row * D;
    for (int d = threadIdx.x; d < D; d += 256)
        out[base + d] = x1[base + d] + g0 * c1[base + d] + g1 * c2[base + d] + g2 * c3[base + d];
}

__global__ void add_kernel(const float* __restrict__ a, const float* __restrict__ b,
                           float* __restrict__ o, long n)
{
    long i = (long)blockIdx.x * blockDim.x + threadIdx.x;
    if (i < n) o[i] = a[i] + b[i];
}

// ---------------------------------------------------------------------------
// Host orchestration
// ---------------------------------------------------------------------------
static void G(hipStream_t st,
              const void* A, int a_is_f32, long az, int lda,
              const bf16_t* Bm, long bz, int ldb, int btrans,
              const float* bias,
              float* Cf, long cfz, int ldcf,
              bf16_t* Cb, long cbz, int ldcb,
              int M, int N, int K, float alpha, int relu, int Z)
{
    dim3 grid((unsigned)(N / BN), (unsigned)(M / BM), (unsigned)Z);
    gemm_wmma<<<grid, dim3(128), 0, st>>>(A, a_is_f32, az, lda, Bm, bz, ldb, btrans,
                                          bias, Cf, cfz, ldcf, Cb, cbz, ldcb,
                                          M, N, K, alpha, relu);
}

extern "C" void kernel_launch(void* const* d_in, const int* in_sizes, int n_in,
                              void* d_out, int out_size, void* d_ws, size_t ws_size,
                              hipStream_t stream)
{
    (void)in_sizes; (void)n_in; (void)out_size; (void)ws_size;

    const float* x     = (const float*)d_in[0];
    const float* src1  = (const float*)d_in[1];
    const float* src11 = (const float*)d_in[2];
    const float* src2  = (const float*)d_in[3];
    const float* src22 = (const float*)d_in[4];
    const float* fedge = (const float*)d_in[5];
    const float* lng   = (const float*)d_in[6];
    const float* lnb   = (const float*)d_in[7];
    const float* wq    = (const float*)d_in[8];
    const float* bq    = (const float*)d_in[9];
    const float* wk    = (const float*)d_in[10];
    const float* bk    = (const float*)d_in[11];
    const float* wv    = (const float*)d_in[12];
    const float* bv    = (const float*)d_in[13];
    const float* wo    = (const float*)d_in[14];
    const float* bo    = (const float*)d_in[15];
    const float* ffw1  = (const float*)d_in[16];
    const float* ffb1  = (const float*)d_in[17];
    const float* ffw2  = (const float*)d_in[18];
    const float* ffb2  = (const float*)d_in[19];
    const float* a1w1  = (const float*)d_in[20];
    const float* a1b1  = (const float*)d_in[21];
    const float* a1w2  = (const float*)d_in[22];
    const float* a1b2  = (const float*)d_in[23];
    const float* a2w1  = (const float*)d_in[24];
    const float* a2b1  = (const float*)d_in[25];
    const float* a2w2  = (const float*)d_in[26];
    const float* a2b2  = (const float*)d_in[27];
    // d_in[28..30]: masks (all-ones / causal tril) -> handled analytically.

    constexpr int Bb = 8, Tt = 512, Dd = 1024, Hh = 16, Ss = 1024, Dff = 2048, Dk = 64;
    const int  NT  = Bb * Tt;      // 4096
    const int  NS  = Bb * Ss;      // 8192
    const long DDl = (long)Dd * Dd;

    char* wsb = (char*)d_ws;
    size_t cur = 0;
    auto alloc = [&](size_t bytes) -> char* {
        char* p = wsb + cur;
        cur += (bytes + 255) & ~(size_t)255;
        return p;
    };
    // bf16 buffers (weights stored TRANSPOSED: [N x K])
    bf16_t* wq_b   = (bf16_t*)alloc(3 * DDl * 2);
    bf16_t* wk_b   = (bf16_t*)alloc(3 * DDl * 2);
    bf16_t* wv_b   = (bf16_t*)alloc(3 * DDl * 2);
    bf16_t* wo_b   = (bf16_t*)alloc(3 * DDl * 2);
    bf16_t* ffw1_b = (bf16_t*)alloc((long)Dd * Dff * 2);
    bf16_t* ffw2_b = (bf16_t*)alloc((long)Dff * Dd * 2);
    bf16_t* a1w1_b = (bf16_t*)alloc((long)2 * Dd * Dd * 2);
    bf16_t* a2w1_b = (bf16_t*)alloc((long)2 * Dd * Dd * 2);
    bf16_t* xn0_b  = (bf16_t*)alloc((long)NT * Dd * 2);
    bf16_t* xn1_b  = (bf16_t*)alloc((long)NT * Dd * 2);
    bf16_t* xn2_b  = (bf16_t*)alloc((long)NT * Dd * 2);
    bf16_t* s1n_b  = (bf16_t*)alloc((long)NS * Dd * 2);
    bf16_t* s11n_b = (bf16_t*)alloc((long)NS * Dd * 2);
    bf16_t* s2n_b  = (bf16_t*)alloc((long)NS * Dd * 2);
    bf16_t* s22n_b = (bf16_t*)alloc((long)NS * Dd * 2);
    bf16_t* qb     = (bf16_t*)alloc((long)NT * Dd * 2);
    bf16_t* kb     = (bf16_t*)alloc((long)NS * Dd * 2);
    bf16_t* vb     = (bf16_t*)alloc((long)NS * Dd * 2);
    bf16_t* ao_b   = (bf16_t*)alloc((long)NT * Dd * 2);
    bf16_t* cc_b   = (bf16_t*)alloc((long)NT * 2 * Dd * 2);
    bf16_t* a1h_b  = (bf16_t*)alloc((long)NT * Dd * 2);
    bf16_t* a2h_b  = (bf16_t*)alloc((long)NT * Dd * 2);
    bf16_t* ffh_b  = (bf16_t*)alloc((long)NT * Dff * 2);
    // f32 buffers
    float* scoresw = (float*)alloc((long)Hh * Tt * Ss * 4);   // reused per-batch
    float* sa_o    = (float*)alloc((long)NT * Dd * 4);
    float* x1      = (float*)alloc((long)NT * Dd * 4);
    float* c1      = (float*)alloc((long)NT * Dd * 4);
    float* c2      = (float*)alloc((long)NT * Dd * 4);
    float* c3      = (float*)alloc((long)NT * Dd * 4);
    float* attn3   = (float*)alloc((long)NT * Ss * 4);
    float* xctx    = (float*)alloc((long)NT * Dd * 4);
    float* ffo     = (float*)alloc((long)NT * Dd * 4);
    float* ig      = (float*)alloc((long)NT * 3 * 4);
    float* fg      = (float*)alloc((long)NT * 4 * 4);

    auto cvtT = [&](const float* in, bf16_t* out, int K, int N) {
        cvt_f32_bf16_t<<<dim3((unsigned)(N / 32), (unsigned)(K / 32)), dim3(256), 0, stream>>>(
            in, out, K, N);
    };
    auto ln = [&](const float* in, int idx, bf16_t* ob, float* of, int rows) {
        layernorm_kernel<<<dim3((unsigned)rows), dim3(256), 0, stream>>>(
            in, lng + (long)idx * Dd, lnb + (long)idx * Dd, ob, of, Dd);
    };

    // ---- phase 0: weight conversion (transposed) + layernorms ----
    for (int l = 0; l < 3; ++l) {
        cvtT(wq + l * DDl, wq_b + l * DDl, Dd, Dd);
        cvtT(wk + l * DDl, wk_b + l * DDl, Dd, Dd);
        cvtT(wv + l * DDl, wv_b + l * DDl, Dd, Dd);
        cvtT(wo + l * DDl, wo_b + l * DDl, Dd, Dd);
    }
    cvtT(ffw1, ffw1_b, Dd, Dff);
    cvtT(ffw2, ffw2_b, Dff, Dd);
    cvtT(a1w1, a1w1_b, 2 * Dd, Dd);
    cvtT(a2w1, a2w1_b, 2 * Dd, Dd);

    ln(x,     0, xn0_b,  nullptr, NT);
    ln(src1,  3, s1n_b,  nullptr, NS);
    ln(src2,  4, s2n_b,  nullptr, NS);
    ln(src11, 5, s11n_b, nullptr, NS);
    ln(src22, 6, s22n_b, nullptr, NS);

    // ---- phase 1: self-attention (causal) ----
    G(stream, xn0_b, 0, 0, Dd, wq_b, 0, Dd, 1, bq, nullptr, 0, 0, qb, 0, Dd, NT, Dd, Dd, 1.f, 0, 1);
    G(stream, xn0_b, 0, 0, Dd, wk_b, 0, Dd, 1, bk, nullptr, 0, 0, kb, 0, Dd, NT, Dd, Dd, 1.f, 0, 1);
    G(stream, xn0_b, 0, 0, Dd, wv_b, 0, Dd, 1, bv, nullptr, 0, 0, vb, 0, Dd, NT, Dd, Dd, 1.f, 0, 1);
    for (int b = 0; b < Bb; ++b) {
        G(stream, qb + (long)b * Tt * Dd, 0, 64, Dd,
          kb + (long)b * Tt * Dd, 64, Dd, 1, nullptr,
          scoresw, (long)Tt * Tt, Tt, nullptr, 0, 0,
          Tt, Tt, Dk, 0.125f, 0, Hh);
        softmax_rows<<<dim3((unsigned)(Hh * Tt)), dim3(256), 0, stream>>>(scoresw, Tt, Tt, 1);
        G(stream, scoresw, 1, (long)Tt * Tt, Tt,
          vb + (long)b * Tt * Dd, 64, Dd, 0, nullptr,
          nullptr, 0, 0, ao_b + (long)b * Tt * Dd, 64, Dd,
          Tt, Dk, Tt, 1.f, 0, Hh);
    }
    G(stream, ao_b, 0, 0, Dd, wo_b, 0, Dd, 1, bo, sa_o, 0, Dd, nullptr, 0, 0, NT, Dd, Dd, 1.f, 0, 1);
    add_kernel<<<dim3((unsigned)(((long)NT * Dd + 255) / 256)), dim3(256), 0, stream>>>(
        x, sa_o, x1, (long)NT * Dd);
    ln(x1, 1, xn1_b, nullptr, NT);

    // ---- phase 2: cross-attention 1 (k=s1n, v=s11n) ----
    G(stream, xn1_b, 0, 0, Dd, wq_b + DDl, 0, Dd, 1, bq + Dd, nullptr, 0, 0, qb, 0, Dd, NT, Dd, Dd, 1.f, 0, 1);
    G(stream, s1n_b, 0, 0, Dd, wk_b + DDl, 0, Dd, 1, bk + Dd, nullptr, 0, 0, kb, 0, Dd, NS, Dd, Dd, 1.f, 0, 1);
    G(stream, s11n_b, 0, 0, Dd, wv_b + DDl, 0, Dd, 1, bv + Dd, nullptr, 0, 0, vb, 0, Dd, NS, Dd, Dd, 1.f, 0, 1);
    for (int b = 0; b < Bb; ++b) {
        G(stream, qb + (long)b * Tt * Dd, 0, 64, Dd,
          kb + (long)b * Ss * Dd, 64, Dd, 1, nullptr,
          scoresw, (long)Tt * Ss, Ss, nullptr, 0, 0,
          Tt, Ss, Dk, 0.125f, 0, Hh);
        softmax_rows<<<dim3((unsigned)(Hh * Tt)), dim3(256), 0, stream>>>(scoresw, Ss, Tt, 0);
        G(stream, scoresw, 1, (long)Tt * Ss, Ss,
          vb + (long)b * Ss * Dd, 64, Dd, 0, nullptr,
          nullptr, 0, 0, ao_b + (long)b * Tt * Dd, 64, Dd,
          Tt, Dk, Ss, 1.f, 0, Hh);
    }
    G(stream, ao_b, 0, 0, Dd, wo_b + DDl, 0, Dd, 1, bo + Dd,
      c1, 0, Dd, cc_b, 0, 2 * Dd, NT, Dd, Dd, 1.f, 0, 1);

    // ---- phase 3: cross-attention 2; probs land in d_out (attn2 output) ----
    float* att2 = (float*)d_out + (long)NT * Dd;   // [B,H,T,S]
    G(stream, xn1_b, 0, 0, Dd, wq_b + 2 * DDl, 0, Dd, 1, bq + 2 * Dd, nullptr, 0, 0, qb, 0, Dd, NT, Dd, Dd, 1.f, 0, 1);
    G(stream, s2n_b, 0, 0, Dd, wk_b + 2 * DDl, 0, Dd, 1, bk + 2 * Dd, nullptr, 0, 0, kb, 0, Dd, NS, Dd, Dd, 1.f, 0, 1);
    G(stream, s22n_b, 0, 0, Dd, wv_b + 2 * DDl, 0, Dd, 1, bv + 2 * Dd, nullptr, 0, 0, vb, 0, Dd, NS, Dd, Dd, 1.f, 0, 1);
    for (int b = 0; b < Bb; ++b) {
        float* sb = att2 + (long)b * Hh * Tt * Ss;
        G(stream, qb + (long)b * Tt * Dd, 0, 64, Dd,
          kb + (long)b * Ss * Dd, 64, Dd, 1, nullptr,
          sb, (long)Tt * Ss, Ss, nullptr, 0, 0,
          Tt, Ss, Dk, 0.125f, 0, Hh);
        softmax_rows<<<dim3((unsigned)(Hh * Tt)), dim3(256), 0, stream>>>(sb, Ss, Tt, 0);
        G(stream, sb, 1, (long)Tt * Ss, Ss,
          vb + (long)b * Ss * Dd, 64, Dd, 0, nullptr,
          nullptr, 0, 0, ao_b + (long)b * Tt * Dd, 64, Dd,
          Tt, Dk, Ss, 1.f, 0, Hh);
    }
    G(stream, ao_b, 0, 0, Dd, wo_b + 2 * DDl, 0, Dd, 1, bo + 2 * Dd,
      c2, 0, Dd, cc_b + Dd, 0, 2 * Dd, NT, Dd, Dd, 1.f, 0, 1);

    // ---- phase 4: gating MLPs ----
    G(stream, cc_b, 0, 0, 2 * Dd, a1w1_b, 0, 2 * Dd, 1, a1b1, nullptr, 0, 0, a1h_b, 0, Dd, NT, Dd, 2 * Dd, 1.f, 1, 1);
    G(stream, cc_b, 0, 0, 2 * Dd, a2w1_b, 0, 2 * Dd, 1, a2b1, nullptr, 0, 0, a2h_b, 0, Dd, NT, Dd, 2 * Dd, 1.f, 1, 1);
    gating_head<<<dim3((unsigned)NT), dim3(128), 0, stream>>>(a1h_b, a1w2, a1b2, ig, Dd, 3);
    gating_head<<<dim3((unsigned)NT), dim3(128), 0, stream>>>(a2h_b, a2w2, a2b2, fg, Dd, 4);

    // ---- phase 5: sequential flow diffusion + c3 ----
    flow_scan<<<dim3((unsigned)Bb), dim3(1024), 0, stream>>>(fedge, fg, attn3, Tt, Ss);
    for (int b = 0; b < Bb; ++b) {
        G(stream, attn3 + (long)b * Tt * Ss, 1, 0, Ss,
          s22n_b + (long)b * Ss * Dd, 0, Dd, 0, nullptr,
          c3 + (long)b * Tt * Dd, 0, Dd, nullptr, 0, 0,
          Tt, Dd, Ss, 1.f, 0, 1);
    }

    // ---- phase 6: gated context combine + FFN ----
    context_combine<<<dim3((unsigned)NT), dim3(256), 0, stream>>>(x1, c1, c2, c3, ig, xctx, Dd);
    ln(xctx, 2, xn2_b, nullptr, NT);
    G(stream, xn2_b, 0, 0, Dd, ffw1_b, 0, Dd, 1, ffb1, nullptr, 0, 0, ffh_b, 0, Dff, NT, Dff, Dd, 1.f, 1, 1);
    G(stream, ffh_b, 0, 0, Dff, ffw2_b, 0, Dff, 1, ffb2, ffo, 0, Dd, nullptr, 0, 0, NT, Dd, Dff, 1.f, 0, 1);
    add_kernel<<<dim3((unsigned)(((long)NT * Dd + 255) / 256)), dim3(256), 0, stream>>>(
        xctx, ffo, (float*)d_out, (long)NT * Dd);
}
